// HaloAttention_64209761075703
// MI455X (gfx1250) — matrix operand order repaired and checked by reading, not compiled, not run on hardware
//
// HaloAttention for MI455X (gfx1250, wave32, WMMA).
// Precision: f16 operands / f32 accumulate via v_wmma_f32_16x16x32_f16.
// ~48 GFLOP vs ~130-250 MB traffic -> f16 WMMA keeps this at the 23.3 TB/s
// HBM floor (f32 WMMA 16x16x4 is 8x lower matrix throughput -> compute bound).
// All four GEMMs (qkv proj, score, ctx, fc) use WMMA; softmax/bias in f32 VALU.
// Weights are converted to f16 once (cvt_w_kernel) so the GEMM kernels do pure
// f16 loads. Attention q/k gathers use GLOBAL_LOAD_ASYNC_TO_LDS_B128 (ASYNCcnt,
// no VGPR round trip); the builtin takes int4* operands per the compiler's
// diagnostic, in AS1 (global) / AS3 (LDS).
#include <hip/hip_runtime.h>

#define DIMC   256
#define HEADS  8
#define DH     32
#define BLK    8
#define HALO   3
#define WIN    14          // BLK + 2*HALO
#define WINSQ  196         // WIN*WIN
#define NKPAD  208         // 13 score N-tiles of 16
#define NKPAD2 224         // 7 ctx K-steps of 32
#define NT_SC  13
#define IMG    128
#define HW     16384       // 128*128

typedef _Float16 h8   __attribute__((ext_vector_type(8)));
typedef _Float16 v16h __attribute__((ext_vector_type(16)));
typedef float    v8f  __attribute__((ext_vector_type(8)));
typedef int      v4i  __attribute__((ext_vector_type(4)));

#if __has_builtin(__builtin_amdgcn_global_load_async_to_lds_b128) && \
    __has_builtin(__builtin_amdgcn_s_wait_asynccnt)
#define USE_ASYNC 1
typedef __attribute__((address_space(1))) v4i gv4i;   // global int4
typedef __attribute__((address_space(3))) v4i lv4i;   // LDS int4
#else
#define USE_ASYNC 0
#endif

static __device__ __forceinline__ v16h cat16(h8 lo, h8 hi) {
  return __builtin_shufflevector(lo, hi, 0,1,2,3,4,5,6,7,8,9,10,11,12,13,14,15);
}
static __device__ __forceinline__ v8f wmma_f16(v16h a, v16h b, v8f c) {
  return __builtin_amdgcn_wmma_f32_16x16x32_f16(false, a, false, b, (short)0, c,
                                                false, false);
}
// 16-byte global->LDS copy: async (ASYNCcnt, no VGPR round trip) when
// available, otherwise a plain load/store pair.
static __device__ __forceinline__ void copy16(const _Float16* g, _Float16* l) {
#if USE_ASYNC
  __builtin_amdgcn_global_load_async_to_lds_b128((gv4i*)g, (lv4i*)l, 0, 0);
#else
  *(h8*)l = *(const h8*)g;
#endif
}
static __device__ __forceinline__ void gather_fence() {
#if USE_ASYNC
  __builtin_amdgcn_s_wait_asynccnt(0);
#endif
  __syncthreads();
}

// ---------------------------------------------------------------------------
// Kernel 0: one-shot weight conversion to f16.
// wqkv16[768][256] = concat(w_q, w_kv); fcw16[256][256] = fc_w.
// ---------------------------------------------------------------------------
__global__ __launch_bounds__(256) void cvt_w_kernel(
    const float* __restrict__ w_q, const float* __restrict__ w_kv,
    const float* __restrict__ fc_w,
    _Float16* __restrict__ wqkv16, _Float16* __restrict__ fcw16)
{
  int i = blockIdx.x * 256 + threadIdx.x;     // 262144 total
  if (i < 65536)        wqkv16[i] = (_Float16)w_q[i];
  else if (i < 196608)  wqkv16[i] = (_Float16)w_kv[i - 65536];
  else                  fcw16[i - 196608] = (_Float16)fc_w[i - 196608];
}

// ---------------------------------------------------------------------------
// Kernel 1: fused QKV projection.  M=768 (wq rows 0..255, wkv rows 256..767),
// N=16384 pixels per batch, K=256.  Outputs channel-last f16 [b][pix][256].
// ---------------------------------------------------------------------------
__global__ __launch_bounds__(256) void proj_kernel(
    const float* __restrict__ x, const _Float16* __restrict__ wqkv,
    _Float16* __restrict__ q_ws, _Float16* __restrict__ k_ws,
    _Float16* __restrict__ v_ws)
{
  __shared__ _Float16 xt[64 * 32];           // x chunk transposed: [pix][c]
  const int tid  = threadIdx.x;
  const int wave = tid >> 5, lane = tid & 31, hi = lane >> 4, lr = lane & 15;
  const int b = blockIdx.z;
  const int pixbase = blockIdx.x * 64;
  const int mwbase  = blockIdx.y * 128 + wave * 16;   // output-channel base

  const _Float16* wrow = wqkv + (size_t)(mwbase + lr) * DIMC;
  const float*    xb   = x + (size_t)b * DIMC * HW;

  v8f acc[4] = {};

  for (int k0 = 0; k0 < DIMC; k0 += 32) {
    __syncthreads();
    for (int i = tid; i < 2048; i += 256) {   // stage [32c][64pix] transposed
      int c = i >> 6, p = i & 63;
      xt[p * 32 + c] = (_Float16)xb[(size_t)(k0 + c) * HW + pixbase + p];
      if (k0 + 32 < DIMC)                     // prefetch next K chunk
        __builtin_prefetch(xb + (size_t)(k0 + 32 + c) * HW + pixbase + p, 0, 0);
    }
    __syncthreads();

    // A: f16 weights, K=hi*16+e (contiguous, matches B)
    v16h a = *(const v16h*)(wrow + k0 + hi * 16);

    #pragma unroll
    for (int nt = 0; nt < 4; ++nt) {          // B: K=hi*16+e, N=lr
      v16h bm = *(const v16h*)(xt + (nt * 16 + lr) * 32 + hi * 16);
      acc[nt] = wmma_f16(a, bm, acc[nt]);
    }
  }

  #pragma unroll
  for (int nt = 0; nt < 4; ++nt) {
    h8 hv;
    #pragma unroll
    for (int vr = 0; vr < 8; ++vr) hv[vr] = (_Float16)acc[nt][vr];
    int o8  = mwbase + hi * 8;                // 8 consecutive channels
    int pix = pixbase + nt * 16 + lr;
    _Float16* dst;
    if (o8 < DIMC)          dst = q_ws + ((size_t)b * HW + pix) * DIMC + o8;
    else if (o8 < 2 * DIMC) dst = k_ws + ((size_t)b * HW + pix) * DIMC + (o8 - DIMC);
    else                    dst = v_ws + ((size_t)b * HW + pix) * DIMC + (o8 - 2 * DIMC);
    *(h8*)dst = hv;
  }
}

// ---------------------------------------------------------------------------
// Kernel 2: halo attention.  One workgroup per (batch, 8x8 block); wave=head.
// LDS: q[64][256] + k[208][256] + v^T[256][224] + bias + p-scratch = 270 KB
// (fits the 320 KB WGP LDS).
// ---------------------------------------------------------------------------
__global__ __launch_bounds__(256) void attn_kernel(
    const _Float16* __restrict__ q_ws, const _Float16* __restrict__ k_ws,
    const _Float16* __restrict__ v_ws, const float* __restrict__ rel_h,
    const float* __restrict__ rel_w, _Float16* __restrict__ ctx_ws)
{
  extern __shared__ __attribute__((aligned(128))) char smem[];
  _Float16* q_lds = (_Float16*)smem;                    // [64][256]   32 KB
  _Float16* k_lds = (_Float16*)(smem + 32768);          // [208][256] 104 KB
  _Float16* v_t   = (_Float16*)(smem + 139264);         // [256][224] 112 KB
  float*    bbase = (float*)(smem + 253952);            // [8][16][28] 14 KB
  _Float16* pbase = (_Float16*)(smem + 268288);         // [8][16][32]  8 KB

  const int tid  = threadIdx.x;
  const int wave = tid >> 5, lane = tid & 31, hi = lane >> 4, lr = lane & 15;
  const int blk = blockIdx.x & 255, b = blockIdx.x >> 8;
  const int h0 = (blk >> 4) * BLK, w0 = (blk & 15) * BLK;

  // gather q block: async global->LDS, 16B contiguous chunks
  for (int i = tid; i < 64 * 32; i += 256) {
    int qi = i >> 5, c8 = (i & 31) * 8;
    int hh = h0 + (qi >> 3), ww = w0 + (qi & 7);
    copy16(q_ws + ((size_t)b * HW + hh * IMG + ww) * DIMC + c8,
           q_lds + qi * DIMC + c8);
  }
  // gather padded k window; invalid chunks zero-filled with disjoint DS stores
  for (int i = tid; i < NKPAD * 32; i += 256) {
    int pos = i >> 5, c8 = (i & 31) * 8;
    bool ok = false;
    int hh = 0, ww = 0;
    if (pos < WINSQ) {
      hh = h0 - HALO + pos / WIN;
      ww = w0 - HALO + pos % WIN;
      ok = (hh >= 0 && hh < IMG && ww >= 0 && ww < IMG);
    }
    if (ok) copy16(k_ws + ((size_t)b * HW + hh * IMG + ww) * DIMC + c8,
                   k_lds + pos * DIMC + c8);
    else    *(h8*)(k_lds + pos * DIMC + c8) = (h8){};
  }
  // gather v transposed -> [channel][key] so ctx B-loads are contiguous
  for (int i = tid; i < NKPAD2 * 32; i += 256) {
    int pos = i >> 5, c8 = (i & 31) * 8;
    h8 v = {};
    if (pos < WINSQ) {
      int hh = h0 - HALO + pos / WIN, ww = w0 - HALO + pos % WIN;
      if (hh >= 0 && hh < IMG && ww >= 0 && ww < IMG)
        v = *(const h8*)(v_ws + ((size_t)b * HW + hh * IMG + ww) * DIMC + c8);
    }
    #pragma unroll
    for (int e = 0; e < 8; ++e) v_t[(c8 + e) * NKPAD2 + pos] = v[e];
  }
  gather_fence();   // s_wait_asynccnt 0 + barrier

  const int hoff = wave * DH;
  float*    bbw  = bbase + wave * 16 * 28;
  _Float16* pscr = pbase + wave * 16 * 32;
  const float scale = 0.17677669529663687f;  // DH^-0.5

  for (int mt = 0; mt < 4; ++mt) {
    // per-query relative-position bias tables: bb[r][0..13]=h(i), [14..27]=w(j)
    for (int e = lane; e < 16 * 28; e += 32) {
      int r = e / 28, c = e % 28;
      int qi = mt * 16 + r, xq = qi >> 3, yq = qi & 7;
      const float* tbl = (c < 14) ? (rel_h + (c - xq + 13) * DH)
                                  : (rel_w + ((c - 14) - yq + 13) * DH);
      const _Float16* qrow = q_lds + qi * DIMC + hoff;
      float acc = 0.f;
      #pragma unroll
      for (int d = 0; d < DH; ++d) acc += (float)qrow[d] * tbl[d];
      bbw[e] = acc;
    }

    // A = q tile (16 queries x K=32)
    const _Float16* qrow = q_lds + (mt * 16 + lr) * DIMC + hoff;
    v16h aq = *(const v16h*)(qrow + hi * 16);

    // score = q @ k^T : 13 WMMAs
    v8f s[NT_SC];
    #pragma unroll
    for (int t = 0; t < NT_SC; ++t) {
      v16h kb = *(const v16h*)(k_lds + (t * 16 + lr) * DIMC + hoff + hi * 16);
      v8f z = {};
      s[t] = wmma_f16(aq, kb, z);
    }

    // scale + bias + mask, row max
    float rmax[8], rsum[8];
    #pragma unroll
    for (int vr = 0; vr < 8; ++vr) rmax[vr] = -1e30f;
    #pragma unroll
    for (int t = 0; t < NT_SC; ++t) {
      int key = t * 16 + lr;                  // N = lane&15
      if (key < WINSQ) {
        int ki = key / WIN, kj = key % WIN;
        #pragma unroll
        for (int vr = 0; vr < 8; ++vr) {      // M = vr + 8*hi
          int rl = vr + hi * 8;
          float v = s[t][vr] * scale + bbw[rl * 28 + ki] + bbw[rl * 28 + 14 + kj];
          s[t][vr] = v;
          rmax[vr] = fmaxf(rmax[vr], v);
        }
      } else {
        #pragma unroll
        for (int vr = 0; vr < 8; ++vr) s[t][vr] = -1e30f;
      }
    }
    #pragma unroll
    for (int vr = 0; vr < 8; ++vr) {          // 16-lane row reduction
      float m = rmax[vr];
      m = fmaxf(m, __shfl_xor(m, 1)); m = fmaxf(m, __shfl_xor(m, 2));
      m = fmaxf(m, __shfl_xor(m, 4)); m = fmaxf(m, __shfl_xor(m, 8));
      rmax[vr] = m; rsum[vr] = 0.f;
    }
    #pragma unroll
    for (int t = 0; t < NT_SC; ++t)
      #pragma unroll
      for (int vr = 0; vr < 8; ++vr) {
        float p = __expf(s[t][vr] - rmax[vr]);
        s[t][vr] = p; rsum[vr] += p;
      }
    #pragma unroll
    for (int vr = 0; vr < 8; ++vr) {
      float sm = rsum[vr];
      sm += __shfl_xor(sm, 1); sm += __shfl_xor(sm, 2);
      sm += __shfl_xor(sm, 4); sm += __shfl_xor(sm, 8);
      rsum[vr] = sm;
    }

    // ctx = p @ v : restripe p through per-wave LDS scratch, 14 WMMAs
    v8f acc0 = {}, acc1 = {};
    #pragma unroll
    for (int ks = 0; ks < 7; ++ks) {
      #pragma unroll
      for (int t2 = 0; t2 < 2; ++t2) {
        int t = ks * 2 + t2;
        #pragma unroll
        for (int vr = 0; vr < 8; ++vr) {
          float p = (t < NT_SC) ? s[t][vr] : 0.f;
          pscr[(vr + hi * 8) * 32 + t2 * 16 + lr] = (_Float16)p;
        }
      }
      const _Float16* prow = pscr + lr * 32;  // per-wave DS ops are in-order
      v16h pa  = *(const v16h*)(prow + hi * 16);
      v16h vb0 = *(const v16h*)(v_t + (hoff + lr)      * NKPAD2 + ks * 32 + hi * 16);
      v16h vb1 = *(const v16h*)(v_t + (hoff + 16 + lr) * NKPAD2 + ks * 32 + hi * 16);
      acc0 = wmma_f16(pa, vb0, acc0);
      acc1 = wmma_f16(pa, vb1, acc1);
    }

    // normalize by row-sum (same lane mapping as C layout) and store f16 ctx
    #pragma unroll
    for (int vr = 0; vr < 8; ++vr) {
      float inv = 1.f / rsum[vr];
      size_t n = (size_t)blockIdx.x * 64 + mt * 16 + vr + hi * 8;
      ctx_ws[n * DIMC + hoff + lr]      = (_Float16)(acc0[vr] * inv);
      ctx_ws[n * DIMC + hoff + 16 + lr] = (_Float16)(acc1[vr] * inv);
    }
  }
}

// ---------------------------------------------------------------------------
// Kernel 3: fc projection.  [65536,256] x fc_w^T [256,256] + bias, scattered
// back into [B, C, H, W] f32.
// ---------------------------------------------------------------------------
__global__ __launch_bounds__(256) void fc_kernel(
    const _Float16* __restrict__ ctx_ws, const _Float16* __restrict__ fcw16,
    const float* __restrict__ fc_b, float* __restrict__ out)
{
  __shared__ _Float16 a_lds[16 * 256];
  const int tid  = threadIdx.x;
  const int wave = tid >> 5, lane = tid & 31, hi = lane >> 4, lr = lane & 15;
  const int nbase = blockIdx.x * 16;

  for (int i = tid; i < 16 * 32; i += 256) {
    int r = i >> 5, c8 = (i & 31) * 8;
    *(h8*)(a_lds + r * 256 + c8) =
        *(const h8*)(ctx_ws + (size_t)(nbase + r) * 256 + c8);
  }
  __syncthreads();

  const int obase = wave * 32;
  v8f acc0 = {}, acc1 = {};
  #pragma unroll
  for (int k0 = 0; k0 < 256; k0 += 32) {
    const _Float16* ar = a_lds + lr * 256 + k0;
    v16h a = *(const v16h*)(ar + hi * 16);
    // B: pre-converted f16 weights, contiguous 32B per lane
    v16h b0 = *(const v16h*)(fcw16 + (size_t)(obase + lr)      * 256 + k0 + hi * 16);
    v16h b1 = *(const v16h*)(fcw16 + (size_t)(obase + 16 + lr) * 256 + k0 + hi * 16);
    acc0 = wmma_f16(a, b0, acc0);
    acc1 = wmma_f16(a, b1, acc1);
  }

  #pragma unroll
  for (int nt = 0; nt < 2; ++nt) {
    v8f acc = nt ? acc1 : acc0;
    int o = obase + nt * 16 + lr;
    float bias = fc_b[o];
    #pragma unroll
    for (int vr = 0; vr < 8; ++vr) {
      int n  = nbase + vr + hi * 8;
      int bb = n >> 14, blk = (n >> 6) & 255, qi = n & 63;
      int hh = (blk >> 4) * 8 + (qi >> 3);
      int ww = (blk & 15) * 8 + (qi & 7);
      out[(((size_t)bb * 256 + o) * 128 + hh) * 128 + ww] = acc[vr] + bias;
    }
  }
}

// ---------------------------------------------------------------------------
extern "C" void kernel_launch(void* const* d_in, const int* in_sizes, int n_in,
                              void* d_out, int out_size, void* d_ws, size_t ws_size,
                              hipStream_t stream) {
  const float* x     = (const float*)d_in[0];
  const float* w_q   = (const float*)d_in[1];
  const float* w_kv  = (const float*)d_in[2];
  const float* fc_w  = (const float*)d_in[3];
  const float* fc_b  = (const float*)d_in[4];
  const float* rel_h = (const float*)d_in[5];
  const float* rel_w = (const float*)d_in[6];
  float* out = (float*)d_out;

  char* ws = (char*)d_ws;
  const size_t SZ = (size_t)4 * HW * DIMC * sizeof(_Float16);  // 32 MB each
  _Float16* q_ws   = (_Float16*)(ws);
  _Float16* k_ws   = (_Float16*)(ws + SZ);
  _Float16* v_ws   = (_Float16*)(ws + 2 * SZ);
  _Float16* ctx_ws = (_Float16*)(ws + 3 * SZ);
  _Float16* wqkv16 = (_Float16*)(ws + 4 * SZ);                 // 384 KB
  _Float16* fcw16  = (_Float16*)(ws + 4 * SZ + 393216);        // 128 KB

  cvt_w_kernel<<<dim3(1024), 256, 0, stream>>>(w_q, w_kv, fc_w, wqkv16, fcw16);
  proj_kernel<<<dim3(256, 6, 4), 256, 0, stream>>>(x, wqkv16, q_ws, k_ws, v_ws);
  attn_kernel<<<dim3(1024), 256, 276480, stream>>>(q_ws, k_ws, v_ws, rel_h, rel_w, ctx_ws);
  fc_kernel<<<dim3(4096), 256, 0, stream>>>(ctx_ws, fcw16, fc_b, out);
}
